// SelfModifyingLinear_90177133347665
// MI455X (gfx1250) — compile-verified, matrix operand order checked by reading, not run on hardware
//
#include <hip/hip_runtime.h>
#include <stdint.h>

#define DIM   2048
#define ROWS  8192          // B*S = 4*2048
#define ETA   0.01f
#define ALPHA 0.95f

typedef __attribute__((ext_vector_type(16))) __bf16          v16bf;
typedef __attribute__((ext_vector_type(8)))  float           v8f;
typedef __attribute__((ext_vector_type(8)))  unsigned short  u16x8;
typedef __attribute__((ext_vector_type(4)))  unsigned short  u16x4;
typedef __attribute__((ext_vector_type(4)))  float           f32x4;

union FragBF {
  v16bf v;
  u16x8 h[2];
  unsigned short s[16];
};

__device__ __forceinline__ unsigned short f2bf(float f) {
  union { float f; unsigned u; } c; c.f = f;
  unsigned u = c.u;
  u += 0x7FFFu + ((u >> 16) & 1u);   // round-to-nearest-even
  return (unsigned short)(u >> 16);
}
__device__ __forceinline__ float bf2f(unsigned short h) {
  union { unsigned u; float f; } c; c.u = ((unsigned)h) << 16;
  return c.f;
}

__device__ __forceinline__ v8f wmma_bf16(v16bf a, v16bf b, v8f c) {
  // (neg_a, A, neg_b, B, c_mod, C, reuse_a, reuse_b)
  return __builtin_amdgcn_wmma_f32_16x16x32_bf16(false, a, false, b,
                                                 (short)0, c, false, false);
}

// Async DMA global -> LDS (GLOBAL_LOAD_ASYNC_TO_LDS_B128, tracked by ASYNCcnt).
// lds32 = DS byte offset (low 32 bits of generic shared pointer), gaddr = global.
__device__ __forceinline__ void async_copy_b128(unsigned lds32,
                                                unsigned long long gaddr) {
  asm volatile("global_load_async_to_lds_b128 %0, %1, off"
               :: "v"(lds32), "v"(gaddr) : "memory");
}
__device__ __forceinline__ void wait_asynccnt0() {
  asm volatile("s_wait_asynccnt 0x0" ::: "memory");
}
__device__ __forceinline__ unsigned lds_off(const void* p) {
  return (unsigned)(uintptr_t)p;
}

// A/B fragment from a row-major [rows][ldst] bf16 LDS tile.
// Per ISA 7.12.2 (16-bit A 16x32): lanes 0-15 hold K=0..7,16..23; lanes 16-31
// hold K=8..15,24..31 -> two 16B reads per lane.
__device__ __forceinline__ v16bf frag_rowmajor(const unsigned short* tile,
                                               int ldst, int rowBase, int lane) {
  int row = rowBase + (lane & 15);
  int off = (lane >> 4) * 8;
  FragBF f;
  f.h[0] = *(const u16x8*)(tile + row * ldst + off);
  f.h[1] = *(const u16x8*)(tile + row * ldst + off + 16);
  return f.v;
}

// Fragment whose M dim is the tile's COLUMN dim and K dim is the tile's ROW dim
// (transposed gather; used for teach^T x style contractions).
__device__ __forceinline__ v16bf frag_transposed(const unsigned short* tile,
                                                 int ldst, int colBase, int lane) {
  int c  = colBase + (lane & 15);
  int kb = (lane >> 4) * 8;
  FragBF f;
#pragma unroll
  for (int j = 0; j < 16; ++j) {
    int k = (j < 8) ? (kb + j) : (16 + kb + (j - 8));
    f.s[j] = tile[k * ldst + c];
  }
  return f.v;
}

// ---------------------------------------------------------------------------
// Kernel 1: Wsum16 = bf16(W + dW), Tdiff16 = bf16(W + dW - tW)
// ---------------------------------------------------------------------------
__global__ __launch_bounds__(256) void prep_weights(
    const float* __restrict__ W, const float* __restrict__ dW,
    const float* __restrict__ tW,
    unsigned short* __restrict__ Wsum16, unsigned short* __restrict__ Tdiff16) {
  int i = (blockIdx.x * 256 + threadIdx.x) * 4;
  f32x4 w  = *(const f32x4*)(W  + i);
  f32x4 d  = *(const f32x4*)(dW + i);
  f32x4 tg = *(const f32x4*)(tW + i);
  u16x4 ws, td;
#pragma unroll
  for (int j = 0; j < 4; ++j) {
    float s = w[j] + d[j];
    ws[j] = f2bf(s);
    td[j] = f2bf(s - tg[j]);
  }
  *(u16x4*)(Wsum16  + i) = ws;
  *(u16x4*)(Tdiff16 + i) = td;
}

// ---------------------------------------------------------------------------
// Kernel 2: X16 = bf16(x)
// ---------------------------------------------------------------------------
__global__ __launch_bounds__(256) void cvt_x(const float* __restrict__ x,
                                             unsigned short* __restrict__ X16) {
  int i = (blockIdx.x * 256 + threadIdx.x) * 4;
  f32x4 v = *(const f32x4*)(x + i);
  u16x4 o;
#pragma unroll
  for (int j = 0; j < 4; ++j) o[j] = f2bf(v[j]);
  *(u16x4*)(X16 + i) = o;
}

// ---------------------------------------------------------------------------
// Kernel 3: fused GEMM: Y = X*Wsum^T (fp32 out), Teach = X*Tdiff^T (bf16 ws)
// 64x64 output tile / block, 8 waves, async-DMA-staged LDS tiles, K step 32.
// ---------------------------------------------------------------------------
#define LDC 40   // padded LDS row stride (elements) for 32-wide K tiles
__global__ __launch_bounds__(256) void gemm_y_teach(
    const unsigned short* __restrict__ X16,
    const unsigned short* __restrict__ Wsum16,
    const unsigned short* __restrict__ Tdiff16,
    float* __restrict__ Yout, unsigned short* __restrict__ Teach16) {
  __shared__ unsigned short sX[64 * LDC];
  __shared__ unsigned short sW[64 * LDC];
  __shared__ unsigned short sT[64 * LDC];

  int m0 = blockIdx.y * 64;
  int n0 = blockIdx.x * 64;
  int t = threadIdx.x, lane = t & 31, w = t >> 5;
  int wm = (w & 3) * 16;          // wave's M sub-tile
  int wn = (w >> 2) * 32;         // wave covers N sub-tiles wn, wn+16

  int lrow = t >> 2;              // 64 rows staged
  int lchunk = (t & 3) * 8;       // 4 x 8 elements per row

  unsigned lx = lds_off(&sX[lrow * LDC + lchunk]);
  unsigned lw = lds_off(&sW[lrow * LDC + lchunk]);
  unsigned lt = lds_off(&sT[lrow * LDC + lchunk]);
  const unsigned short* gx = &X16[(size_t)(m0 + lrow) * DIM + lchunk];
  const unsigned short* gw = &Wsum16[(size_t)(n0 + lrow) * DIM + lchunk];
  const unsigned short* gt = &Tdiff16[(size_t)(n0 + lrow) * DIM + lchunk];

  v8f cy0 = {}, cy1 = {}, ct0 = {}, ct1 = {};

  for (int k0 = 0; k0 < DIM; k0 += 32) {
    async_copy_b128(lx, (unsigned long long)(uintptr_t)(gx + k0));
    async_copy_b128(lw, (unsigned long long)(uintptr_t)(gw + k0));
    async_copy_b128(lt, (unsigned long long)(uintptr_t)(gt + k0));
    if (k0 + 32 < DIM) {  // uniform branch: prefetch next K tile
      __builtin_prefetch(gx + k0 + 32, 0, 1);
      __builtin_prefetch(gw + k0 + 32, 0, 1);
      __builtin_prefetch(gt + k0 + 32, 0, 1);
    }
    wait_asynccnt0();
    __syncthreads();

    v16bf a   = frag_rowmajor(sX, LDC, wm, lane);
    v16bf bw0 = frag_rowmajor(sW, LDC, wn, lane);
    v16bf bw1 = frag_rowmajor(sW, LDC, wn + 16, lane);
    v16bf bt0 = frag_rowmajor(sT, LDC, wn, lane);
    v16bf bt1 = frag_rowmajor(sT, LDC, wn + 16, lane);

    cy0 = wmma_bf16(a, bw0, cy0);
    cy1 = wmma_bf16(a, bw1, cy1);
    ct0 = wmma_bf16(a, bt0, ct0);
    ct1 = wmma_bf16(a, bt1, ct1);
    __syncthreads();
  }

  // C layout: VGPR v, lanes 0-15 -> M=v, lanes 16-31 -> M=v+8; N = lane&15
  int col = n0 + wn + (lane & 15);
  int rowB = m0 + wm + ((lane >> 4) << 3);
#pragma unroll
  for (int v = 0; v < 8; ++v) {
    size_t r = (size_t)(rowB + v) * DIM;
    Yout[r + col]      = cy0[v];
    Yout[r + col + 16] = cy1[v];
    Teach16[r + col]      = f2bf(ct0[v]);
    Teach16[r + col + 16] = f2bf(ct1[v]);
  }
}

// ---------------------------------------------------------------------------
// Kernel 4: per-row L2 norm of teach
// ---------------------------------------------------------------------------
__global__ __launch_bounds__(256) void row_norm(
    const unsigned short* __restrict__ Teach16, float* __restrict__ rowNorm) {
  __shared__ float red[256];
  int r = blockIdx.x, t = threadIdx.x;
  const unsigned short* row = Teach16 + (size_t)r * DIM;
  u16x8 v = *(const u16x8*)(row + t * 8);
  float s = 0.f;
#pragma unroll
  for (int j = 0; j < 8; ++j) {
    float f = bf2f(v[j]);
    s += f * f;
  }
  red[t] = s;
  __syncthreads();
  for (int off = 128; off > 0; off >>= 1) {
    if (t < off) red[t] += red[t + off];
    __syncthreads();
  }
  if (t == 0) rowNorm[r] = sqrtf(red[0]);
}

// ---------------------------------------------------------------------------
// Kernel 5: surprise = mean(rowNorm); emit gated update coefficients
// ---------------------------------------------------------------------------
__global__ __launch_bounds__(256) void gate_kernel(
    const float* __restrict__ rowNorm, float* __restrict__ coefs) {
  __shared__ float red[256];
  int t = threadIdx.x;
  float s = 0.f;
  for (int i = t; i < ROWS; i += 256) s += rowNorm[i];
  red[t] = s;
  __syncthreads();
  for (int off = 128; off > 0; off >>= 1) {
    if (t < off) red[t] += red[t + off];
    __syncthreads();
  }
  if (t == 0) {
    float surprise = red[0] * (1.0f / (float)ROWS);
    bool g = surprise > 1.0f;
    coefs[0] = g ? ALPHA : 1.0f;                    // multiplies old delta_W
    coefs[1] = g ? (ETA / (float)ROWS) : 0.0f;      // multiplies raw teach^T x
  }
}

// ---------------------------------------------------------------------------
// Kernel 6: delta = Teach^T * X (K over rows, transposed LDS fragments),
// fused with gated update: dWout = cA*dW + cB*delta_raw
// ---------------------------------------------------------------------------
#define LDE 72   // padded LDS row stride for 64-wide tiles
__global__ __launch_bounds__(256) void gemm_delta_update(
    const unsigned short* __restrict__ Teach16,
    const unsigned short* __restrict__ X16,
    const float* __restrict__ dW, const float* __restrict__ coefs,
    float* __restrict__ dWout) {
  __shared__ unsigned short sT[32 * LDE];
  __shared__ unsigned short sX[32 * LDE];

  int o0 = blockIdx.y * 64;       // output row (O) tile
  int i0 = blockIdx.x * 64;       // output col (I) tile
  int t = threadIdx.x, lane = t & 31, w = t >> 5;
  int wm = (w & 3) * 16;
  int wn = (w >> 2) * 32;

  int lrow = t >> 3;              // 32 K-rows staged
  int lchunk = (t & 7) * 8;       // 8 x 8 elements per row (64 cols)

  unsigned ltt = lds_off(&sT[lrow * LDE + lchunk]);
  unsigned lxx = lds_off(&sX[lrow * LDE + lchunk]);
  const unsigned short* gt = &Teach16[(size_t)lrow * DIM + o0 + lchunk];
  const unsigned short* gx = &X16[(size_t)lrow * DIM + i0 + lchunk];

  v8f c0 = {}, c1 = {};

  for (int r0 = 0; r0 < ROWS; r0 += 32) {
    size_t roff = (size_t)r0 * DIM;
    async_copy_b128(ltt, (unsigned long long)(uintptr_t)(gt + roff));
    async_copy_b128(lxx, (unsigned long long)(uintptr_t)(gx + roff));
    if (r0 + 32 < ROWS) {
      __builtin_prefetch(gt + roff + (size_t)32 * DIM, 0, 1);
      __builtin_prefetch(gx + roff + (size_t)32 * DIM, 0, 1);
    }
    wait_asynccnt0();
    __syncthreads();

    v16bf a  = frag_transposed(sT, LDE, wm, lane);       // A[m=o][k=r]
    v16bf b0 = frag_transposed(sX, LDE, wn, lane);       // B[n=i][k=r]
    v16bf b1 = frag_transposed(sX, LDE, wn + 16, lane);

    c0 = wmma_bf16(a, b0, c0);
    c1 = wmma_bf16(a, b1, c1);
    __syncthreads();
  }

  float cA = coefs[0];
  float cB = coefs[1];
  int col = i0 + wn + (lane & 15);
  int rowB = o0 + wm + ((lane >> 4) << 3);
#pragma unroll
  for (int v = 0; v < 8; ++v) {
    size_t r = (size_t)(rowB + v) * DIM;
    dWout[r + col]      = cA * dW[r + col]      + cB * c0[v];
    dWout[r + col + 16] = cA * dW[r + col + 16] + cB * c1[v];
  }
}

// ---------------------------------------------------------------------------
extern "C" void kernel_launch(void* const* d_in, const int* in_sizes, int n_in,
                              void* d_out, int out_size, void* d_ws, size_t ws_size,
                              hipStream_t stream) {
  const float* x  = (const float*)d_in[0];   // [8192, 2048]
  const float* W  = (const float*)d_in[1];   // [2048, 2048]
  const float* tW = (const float*)d_in[2];   // [2048, 2048]
  const float* dW = (const float*)d_in[3];   // [2048, 2048]

  float* y     = (float*)d_out;                       // [8192, 2048]
  float* dWout = y + (size_t)ROWS * DIM;              // [2048, 2048]

  char* ws = (char*)d_ws;
  unsigned short* Wsum16  = (unsigned short*)(ws);                          // 8 MB
  unsigned short* Tdiff16 = (unsigned short*)(ws + ((size_t)8  << 20));     // 8 MB
  unsigned short* X16     = (unsigned short*)(ws + ((size_t)16 << 20));     // 32 MB
  unsigned short* Teach16 = (unsigned short*)(ws + ((size_t)48 << 20));     // 32 MB
  float* rowNorm          = (float*)(ws + ((size_t)80 << 20));              // 32 KB
  float* coefs            = (float*)(ws + ((size_t)80 << 20) + 32768);

  prep_weights<<<(DIM * DIM) / (256 * 4), 256, 0, stream>>>(W, dW, tW, Wsum16, Tdiff16);
  cvt_x<<<((size_t)ROWS * DIM) / (256 * 4), 256, 0, stream>>>(x, X16);

  dim3 gC(DIM / 64, ROWS / 64);   // 32 x 128
  gemm_y_teach<<<gC, 256, 0, stream>>>(X16, Wsum16, Tdiff16, y, Teach16);

  row_norm<<<ROWS, 256, 0, stream>>>(Teach16, rowNorm);
  gate_kernel<<<1, 256, 0, stream>>>(rowNorm, coefs);

  dim3 gE(DIM / 64, DIM / 64);    // 32 x 32
  gemm_delta_update<<<gE, 256, 0, stream>>>(Teach16, X16, dW, coefs, dWout);
}
